// DocFormer_36438502539336
// MI455X (gfx1250) — matrix operand order repaired and checked by reading, not compiled, not run on hardware
//
#include <hip/hip_runtime.h>

// ---------------------------------------------------------------------------
// DocFormer dual-stream attention for gfx1250 (MI455X), wave32 + WMMA bf16.
// ---------------------------------------------------------------------------

typedef __bf16 bf16;
typedef __attribute__((ext_vector_type(16))) __bf16 v16bf;
typedef __attribute__((ext_vector_type(8)))  __bf16 bf16x8;
typedef __attribute__((ext_vector_type(8)))  float  v8f;

#define EMBED 768
#define HEADS 12
#define HEAD_DIM 64
#define SEQ 512
#define BATCH 8
#define NROW 4096              // BATCH*SEQ
#define INV_SCALE 0.03608439182435161f   // 1/sqrt(768)

__device__ inline v8f v8f_zero() {
  v8f z;
#pragma unroll
  for (int i = 0; i < 8; ++i) z[i] = 0.0f;
  return z;
}

__device__ inline v8f wmma_bf16(v16bf a, v16bf b, v8f c) {
  return __builtin_amdgcn_wmma_f32_16x16x32_bf16(false, a, false, b,
                                                 (short)0, c, false, false);
}

union V16U { v16bf v; bf16x8 h[2]; };

// A-matrix 16x32 bf16 fragment (per ISA table): lane holds row = lane&15,
// elements 0..7  -> K = 8*half + i, elements 8..15 -> K = 16 + 8*half + (i-8).
// Two contiguous 8-element (16B) runs.
__device__ inline v16bf load_a_frag(const bf16* rowptr, int kb, int hf) {
  V16U u;
  u.h[0] = *(const bf16x8*)(rowptr + kb + 8 * hf);
  u.h[1] = *(const bf16x8*)(rowptr + kb + 16 + 8 * hf);
  return u.v;
}

// B-matrix 32x16 bf16 fragment: lane holds col = lane&15,
// elements 0..7 -> K = 8*half + i, elements 8..15 -> K = 16 + 8*half + (i-8).
// colptr points to the start of this column's K-contiguous data.
__device__ inline v16bf load_b_frag(const bf16* colptr, int kb, int hf) {
  V16U u;
  u.h[0] = *(const bf16x8*)(colptr + kb + 8 * hf);
  u.h[1] = *(const bf16x8*)(colptr + kb + 16 + 8 * hf);
  return u.v;
}

// A-frag from f32 source (convert to bf16 on the fly).
__device__ inline v16bf load_a_frag_f32(const float* rowptr, int kb, int hf) {
  v16bf r;
  const float* p0 = rowptr + kb + 8 * hf;
  const float* p1 = rowptr + kb + 16 + 8 * hf;
#pragma unroll
  for (int i = 0; i < 8; ++i) {
    r[i]     = (bf16)p0[i];
    r[8 + i] = (bf16)p1[i];
  }
  return r;
}

__device__ inline int iclamp(int x, int lo, int hi) {
  return x < lo ? lo : (x > hi ? hi : x);
}

// ---------------------------------------------------------------------------
// Kernel 1: projection GEMM  Y = X @ W + b  (4096x768 @ 768x768), bf16 out.
// One 16x16 output tile per wave; K in chunks of 32 via v_wmma_f32_16x16x32_bf16.
// W chunk staged transposed (col-major-in-K) into LDS as bf16 so B-fragments
// are two ds_load_b128 per lane.
// TRANSPOSE_V=1: store Y as [h][b][d][t] (for V) with one packed
// global_store_b128 per lane; TRANSPOSE_V=0: [b*t][h*64+d] row-major.
// ---------------------------------------------------------------------------
template <int TRANSPOSE_V>
__global__ __launch_bounds__(128) void proj_kernel(
    const float* __restrict__ X, const float* __restrict__ W,
    const float* __restrict__ bias, bf16* __restrict__ Y) {
  __shared__ __align__(16) bf16 wt[4][16 * 32];   // per-wave [col][k] tile
  const int tid = threadIdx.x;
  const int wid = tid >> 5;
  const int ln  = tid & 31;
  const int hf  = ln >> 4;
  const int col = ln & 15;
  const int ntile = blockIdx.x * 4 + wid;    // 0..47
  const int mtile = blockIdx.y;              // 0..255
  const int ncol0 = ntile * 16;
  const int arow  = mtile * 16 + col;
  const float* xrow = X + (size_t)arow * EMBED;
  bf16* lds = wt[wid];

  v8f acc = v8f_zero();
  for (int kb = 0; kb < EMBED; kb += 32) {
    // stage 32x16 W chunk transposed into LDS (bf16)
#pragma unroll
    for (int i = 0; i < 16; ++i) {
      int kl = i * 2 + hf;                  // 0..31 across the half
      lds[col * 32 + kl] = (bf16)W[(size_t)(kb + kl) * EMBED + ncol0 + col];
    }
    v16bf af = load_a_frag_f32(xrow, kb, hf);
    v16bf bfg = load_b_frag(lds + col * 32, 0, hf);
    acc = wmma_bf16(af, bfg, acc);
  }

  // C/D layout: element r -> row = r + 8*half, col = lane&15
  const int ocol = ncol0 + col;
  const float bv = bias[ocol];
  if (TRANSPOSE_V) {
    // rows r=0..7 are consecutive t within one batch; pack 8 bf16 -> b128
    bf16x8 pk;
#pragma unroll
    for (int r = 0; r < 8; ++r) pk[r] = (bf16)(acc[r] + bv);
    int baserow = mtile * 16 + 8 * hf;       // first of 8 consecutive rows
    int h = ocol >> 6, d = ocol & 63;
    int b = baserow >> 9, t0 = baserow & 511;
    *(bf16x8*)(Y + (((size_t)(h * BATCH + b) * HEAD_DIM + d) << 9) + t0) = pk;
  } else {
#pragma unroll
    for (int r = 0; r < 8; ++r) {
      int orow = mtile * 16 + r + 8 * hf;
      Y[(size_t)orow * EMBED + ocol] = (bf16)(acc[r] + bv);
    }
  }
}

// ---------------------------------------------------------------------------
// Kernel 2: rel-pos projections.
//   rq[s,h,b,t,j] = dot(q_s[h,b,t,:], table[j,:])   (same for rk with k)
// Collapses the O(L^2 d) rel einsums into gathers of these 17-wide tables.
// ---------------------------------------------------------------------------
__global__ __launch_bounds__(256) void relproj_kernel(
    const bf16* __restrict__ q_t, const bf16* __restrict__ k_t,
    const bf16* __restrict__ q_i, const bf16* __restrict__ k_i,
    const float* __restrict__ table, float* __restrict__ rq,
    float* __restrict__ rk) {
  int idx = blockIdx.x * blockDim.x + threadIdx.x;
  if (idx >= 2 * HEADS * BATCH * SEQ) return;
  int t = idx & 511;
  int r = idx >> 9;
  int b = r & 7; r >>= 3;
  int h = r % HEADS;
  int s = r / HEADS;
  const bf16* qa = (s ? q_i : q_t) + ((size_t)(b * SEQ + t) * EMBED + h * HEAD_DIM);
  const bf16* ka = (s ? k_i : k_t) + ((size_t)(b * SEQ + t) * EMBED + h * HEAD_DIM);
  for (int j = 0; j < 17; ++j) {
    const float* tb = table + j * HEAD_DIM;
    float sq = 0.0f, sk = 0.0f;
#pragma unroll 8
    for (int d = 0; d < HEAD_DIM; ++d) {
      sq += (float)qa[d] * tb[d];
      sk += (float)ka[d] * tb[d];
    }
    rq[(size_t)idx * 17 + j] = sq;
    rk[(size_t)idx * 17 + j] = sk;
  }
}

// ---------------------------------------------------------------------------
// Kernel 3: flash attention, both streams accumulated into ctx.
// One wave handles a 16-query tile for one (h,b). Key blocks of 32 ->
// two 16x16 score accumulators (QK^T + spatial QsKs^T, 2 d-chunks each),
// scale + rel bias, online softmax, P through LDS (C->A relayout), P@V.
// ---------------------------------------------------------------------------
__global__ __launch_bounds__(128) void attn_kernel(
    const bf16* __restrict__ q_t, const bf16* __restrict__ k_t,
    const bf16* __restrict__ vt_t, const bf16* __restrict__ q_i,
    const bf16* __restrict__ k_i, const bf16* __restrict__ vt_i,
    const bf16* __restrict__ qs_t, const bf16* __restrict__ ks_t,
    const bf16* __restrict__ qs_i, const bf16* __restrict__ ks_i,
    const float* __restrict__ rq, const float* __restrict__ rk,
    bf16* __restrict__ ctx) {
  __shared__ __align__(16) bf16 pbuf[4][16 * 32];
  const int tid = threadIdx.x;
  const int wid = tid >> 5;
  const int ln  = tid & 31;
  const int hf  = ln >> 4;
  const int col = ln & 15;
  const int h = blockIdx.z;
  const int b = blockIdx.y;
  const int qtile = blockIdx.x * 4 + wid;            // 0..31
  const int qrow  = qtile * 16 + col;                // A-frag row for this lane
  const size_t hb_off = (size_t)b * SEQ * EMBED + h * HEAD_DIM;
  bf16* lds = pbuf[wid];

  v8f cx0 = v8f_zero(), cx1 = v8f_zero(), cx2 = v8f_zero(), cx3 = v8f_zero();

  for (int s = 0; s < 2; ++s) {
    const bf16* qarr  = s ? q_i  : q_t;
    const bf16* karr  = s ? k_i  : k_t;
    const bf16* vtarr = s ? vt_i : vt_t;
    const bf16* qsarr = s ? qs_i : qs_t;
    const bf16* ksarr = s ? ks_i : ks_t;
    const float* rqs = rq + (size_t)(((s * HEADS + h) * BATCH + b) * SEQ) * 17;
    const float* rks = rk + (size_t)(((s * HEADS + h) * BATCH + b) * SEQ) * 17;
    const bf16* vt_hb = vtarr + ((size_t)(h * BATCH + b) * HEAD_DIM << 9);

    // Q and Qs A-fragments for the whole tile (d = 64 -> two K-chunks of 32)
    const bf16* qp  = qarr  + hb_off + (size_t)qrow * EMBED;
    const bf16* qsp = qsarr + hb_off + (size_t)qrow * EMBED;
    v16bf aq0 = load_a_frag(qp, 0, hf);
    v16bf aq1 = load_a_frag(qp, 32, hf);
    v16bf as0 = load_a_frag(qsp, 0, hf);
    v16bf as1 = load_a_frag(qsp, 32, hf);

    v8f o0 = v8f_zero(), o1 = v8f_zero(), o2 = v8f_zero(), o3 = v8f_zero();
    float m[8], l[8];
#pragma unroll
    for (int r = 0; r < 8; ++r) { m[r] = -3.0e38f; l[r] = 0.0f; }

    for (int kb = 0; kb < SEQ; kb += 32) {
      // --- scores: two 16x16 tiles (keys kb..kb+15 and kb+16..kb+31) ---
      const bf16* kc0  = karr  + hb_off + (size_t)(kb + col) * EMBED;
      const bf16* kc1  = karr  + hb_off + (size_t)(kb + 16 + col) * EMBED;
      const bf16* ksc0 = ksarr + hb_off + (size_t)(kb + col) * EMBED;
      const bf16* ksc1 = ksarr + hb_off + (size_t)(kb + 16 + col) * EMBED;

      v8f s0 = v8f_zero(), s1 = v8f_zero();
      s0 = wmma_bf16(aq0, load_b_frag(kc0, 0, hf), s0);
      s0 = wmma_bf16(aq1, load_b_frag(kc0, 32, hf), s0);
      s0 = wmma_bf16(as0, load_b_frag(ksc0, 0, hf), s0);
      s0 = wmma_bf16(as1, load_b_frag(ksc0, 32, hf), s0);
      s1 = wmma_bf16(aq0, load_b_frag(kc1, 0, hf), s1);
      s1 = wmma_bf16(aq1, load_b_frag(kc1, 32, hf), s1);
      s1 = wmma_bf16(as0, load_b_frag(ksc1, 0, hf), s1);
      s1 = wmma_bf16(as1, load_b_frag(ksc1, 32, hf), s1);

      // --- scale + relative-position bias, then online softmax ---
#pragma unroll
      for (int r = 0; r < 8; ++r) {
        int rg  = qtile * 16 + r + 8 * hf;        // global query row
        int cg0 = kb + col;                       // global key cols
        int cg1 = kb + 16 + col;
        int j0 = iclamp(cg0 - rg, -8, 8) + 8;
        int j1 = iclamp(cg1 - rg, -8, 8) + 8;
        float v0 = s0[r] * INV_SCALE + rks[(size_t)cg0 * 17 + j0] + rqs[(size_t)rg * 17 + j0];
        float v1 = s1[r] * INV_SCALE + rks[(size_t)cg1 * 17 + j1] + rqs[(size_t)rg * 17 + j1];

        float mx = fmaxf(v0, v1);
        mx = fmaxf(mx, __shfl_xor(mx, 1, 32));
        mx = fmaxf(mx, __shfl_xor(mx, 2, 32));
        mx = fmaxf(mx, __shfl_xor(mx, 4, 32));
        mx = fmaxf(mx, __shfl_xor(mx, 8, 32));
        float mnew  = fmaxf(m[r], mx);
        float alpha = __expf(m[r] - mnew);
        m[r] = mnew;
        float e0 = __expf(v0 - mnew);
        float e1 = __expf(v1 - mnew);
        float rs = e0 + e1;
        rs += __shfl_xor(rs, 1, 32);
        rs += __shfl_xor(rs, 2, 32);
        rs += __shfl_xor(rs, 4, 32);
        rs += __shfl_xor(rs, 8, 32);
        l[r] = l[r] * alpha + rs;
        o0[r] = o0[r] * alpha;
        o1[r] = o1[r] * alpha;
        o2[r] = o2[r] * alpha;
        o3[r] = o3[r] * alpha;

        // P -> LDS (bf16) for C->A fragment relayout
        int prow = r + 8 * hf;
        lds[prow * 32 + col]      = (bf16)e0;
        lds[prow * 32 + 16 + col] = (bf16)e1;
      }

      // --- P (16x32) @ V (32x64): A-frag from LDS, B-frags from V^T global ---
      v16bf ap = load_a_frag(lds + col * 32, 0, hf);
      o0 = wmma_bf16(ap, load_b_frag(vt_hb + ((size_t)(0 * 16 + col) << 9), kb, hf), o0);
      o1 = wmma_bf16(ap, load_b_frag(vt_hb + ((size_t)(1 * 16 + col) << 9), kb, hf), o1);
      o2 = wmma_bf16(ap, load_b_frag(vt_hb + ((size_t)(2 * 16 + col) << 9), kb, hf), o2);
      o3 = wmma_bf16(ap, load_b_frag(vt_hb + ((size_t)(3 * 16 + col) << 9), kb, hf), o3);
    }

    // per-stream normalization, accumulate into ctx
#pragma unroll
    for (int r = 0; r < 8; ++r) {
      float inv = 1.0f / l[r];
      cx0[r] += o0[r] * inv;
      cx1[r] += o1[r] * inv;
      cx2[r] += o2[r] * inv;
      cx3[r] += o3[r] * inv;
    }
  }

  // store ctx: [b*SEQ+t][h*64+d] bf16
#pragma unroll
  for (int r = 0; r < 8; ++r) {
    int t = qtile * 16 + r + 8 * hf;
    size_t ro = (size_t)(b * SEQ + t) * EMBED + h * HEAD_DIM;
    ctx[ro + 0 * 16 + col] = (bf16)cx0[r];
    ctx[ro + 1 * 16 + col] = (bf16)cx1[r];
    ctx[ro + 2 * 16 + col] = (bf16)cx2[r];
    ctx[ro + 3 * 16 + col] = (bf16)cx3[r];
  }
}

// ---------------------------------------------------------------------------
// Kernel 4: output projection  out = ctx @ Wo + bo  (f32 output).
// ---------------------------------------------------------------------------
__global__ __launch_bounds__(128) void outproj_kernel(
    const bf16* __restrict__ ctx, const float* __restrict__ W,
    const float* __restrict__ bias, float* __restrict__ out) {
  __shared__ __align__(16) bf16 wt[4][16 * 32];
  const int tid = threadIdx.x;
  const int wid = tid >> 5;
  const int ln  = tid & 31;
  const int hf  = ln >> 4;
  const int col = ln & 15;
  const int ntile = blockIdx.x * 4 + wid;
  const int mtile = blockIdx.y;
  const int ncol0 = ntile * 16;
  const int arow  = mtile * 16 + col;
  const bf16* crow = ctx + (size_t)arow * EMBED;
  bf16* lds = wt[wid];

  v8f acc = v8f_zero();
  for (int kb = 0; kb < EMBED; kb += 32) {
#pragma unroll
    for (int i = 0; i < 16; ++i) {
      int kl = i * 2 + hf;
      lds[col * 32 + kl] = (bf16)W[(size_t)(kb + kl) * EMBED + ncol0 + col];
    }
    v16bf af  = load_a_frag(crow, kb, hf);
    v16bf bfg = load_b_frag(lds + col * 32, 0, hf);
    acc = wmma_bf16(af, bfg, acc);
  }
  const int ocol = ncol0 + col;
  const float bv = bias[ocol];
#pragma unroll
  for (int r = 0; r < 8; ++r) {
    int orow = mtile * 16 + r + 8 * hf;
    out[(size_t)orow * EMBED + ocol] = acc[r] + bv;
  }
}

// ---------------------------------------------------------------------------
// Launcher. Workspace layout (bytes):
//   [0 .. 11*PROJ*2)   : 11 bf16 buffers of 4096*768 each
//                        0:q_t 1:k_t 2:vT_t 3:q_i 4:k_i 5:vT_i
//                        6:qs_t 7:ks_t 8:qs_i 9:ks_i 10:ctx
//   then rq, rk        : each 2*12*8*512*17 f32
// Total ~83 MB.
// ---------------------------------------------------------------------------
extern "C" void kernel_launch(void* const* d_in, const int* in_sizes, int n_in,
                              void* d_out, int out_size, void* d_ws,
                              size_t ws_size, hipStream_t stream) {
  const float* text   = (const float*)d_in[0];
  const float* img    = (const float*)d_in[1];
  const float* text_s = (const float*)d_in[2];
  const float* img_s  = (const float*)d_in[3];
  const float* Wq_t = (const float*)d_in[4];  const float* bq_t = (const float*)d_in[5];
  const float* Wk_t = (const float*)d_in[6];  const float* bk_t = (const float*)d_in[7];
  const float* Wv_t = (const float*)d_in[8];  const float* bv_t = (const float*)d_in[9];
  const float* Wq_i = (const float*)d_in[10]; const float* bq_i = (const float*)d_in[11];
  const float* Wk_i = (const float*)d_in[12]; const float* bk_i = (const float*)d_in[13];
  const float* Wv_i = (const float*)d_in[14]; const float* bv_i = (const float*)d_in[15];
  const float* Wq_s = (const float*)d_in[16]; const float* bq_s = (const float*)d_in[17];
  const float* Wk_s = (const float*)d_in[18]; const float* bk_s = (const float*)d_in[19];
  const float* Wo   = (const float*)d_in[20]; const float* bo   = (const float*)d_in[21];
  const float* rel_table = (const float*)d_in[22];

  const size_t PROJ = (size_t)NROW * EMBED;          // elements per buffer
  char* ws = (char*)d_ws;
  bf16* buf[11];
  for (int i = 0; i < 11; ++i) buf[i] = (bf16*)(ws + (size_t)i * PROJ * 2);
  bf16 *q_t = buf[0], *k_t = buf[1], *vT_t = buf[2];
  bf16 *q_i = buf[3], *k_i = buf[4], *vT_i = buf[5];
  bf16 *qs_t = buf[6], *ks_t = buf[7], *qs_i = buf[8], *ks_i = buf[9];
  bf16* ctxb = buf[10];
  const size_t RELN = (size_t)2 * HEADS * BATCH * SEQ * 17;
  float* rqb = (float*)(ws + (size_t)11 * PROJ * 2);
  float* rkb = rqb + RELN;

  dim3 pb(128), pg(12, 256);
  proj_kernel<0><<<pg, pb, 0, stream>>>(text,   Wq_t, bq_t, q_t);
  proj_kernel<0><<<pg, pb, 0, stream>>>(text,   Wk_t, bk_t, k_t);
  proj_kernel<1><<<pg, pb, 0, stream>>>(text,   Wv_t, bv_t, vT_t);
  proj_kernel<0><<<pg, pb, 0, stream>>>(img,    Wq_i, bq_i, q_i);
  proj_kernel<0><<<pg, pb, 0, stream>>>(img,    Wk_i, bk_i, k_i);
  proj_kernel<1><<<pg, pb, 0, stream>>>(img,    Wv_i, bv_i, vT_i);
  proj_kernel<0><<<pg, pb, 0, stream>>>(text_s, Wq_s, bq_s, qs_t);
  proj_kernel<0><<<pg, pb, 0, stream>>>(text_s, Wk_s, bk_s, ks_t);
  proj_kernel<0><<<pg, pb, 0, stream>>>(img_s,  Wq_s, bq_s, qs_i);
  proj_kernel<0><<<pg, pb, 0, stream>>>(img_s,  Wk_s, bk_s, ks_i);

  int relThreads = 2 * HEADS * BATCH * SEQ;
  relproj_kernel<<<(relThreads + 255) / 256, 256, 0, stream>>>(
      q_t, k_t, q_i, k_i, rel_table, rqb, rkb);

  attn_kernel<<<dim3(8, BATCH, HEADS), 128, 0, stream>>>(
      q_t, k_t, vT_t, q_i, k_i, vT_i, qs_t, ks_t, qs_i, ks_i, rqb, rkb, ctxb);

  outproj_kernel<<<dim3(12, 256), 128, 0, stream>>>(ctxb, Wo, bo, (float*)d_out);
}
